// EncodeProcessDecode_28862180229753
// MI455X (gfx1250) — matrix operand order repaired and checked by reading, not compiled, not run on hardware
//
#include <hip/hip_runtime.h>

// ---------------------------------------------------------------------------
// MI455X (gfx1250) Encode-Process-Decode GNN, bf16 WMMA implementation.
// ---------------------------------------------------------------------------

typedef unsigned short bf16_t;                                // raw bf16 bits
typedef __attribute__((ext_vector_type(16))) __bf16 v16bf;    // WMMA A/B frag
typedef __attribute__((ext_vector_type(8)))  float  v8f;      // WMMA C/D frag

static constexpr int NN = 50000;    // nodes
static constexpr int NE = 800000;   // edges
static constexpr int ROWP = 104;    // padded LDS row (bf16 elems), 208B

union ABFrag { v16bf v; uint4 q[2]; };

struct Seg {
    const bf16_t* base;   // row-major bf16, row stride = stride elems
    const int*    idx;    // optional row indirection (nullptr = identity)
    int           stride; // 0 => broadcast row 0
};

__device__ __forceinline__ bf16_t f2bf(float f) {
    union { float f; unsigned u; } v; v.f = f;
    unsigned r = v.u + 0x7FFFu + ((v.u >> 16) & 1u);   // RNE
    return (bf16_t)(r >> 16);
}
__device__ __forceinline__ float bf2f(bf16_t b) {
    union { unsigned u; float f; } v; v.u = ((unsigned)b) << 16;
    return v.f;
}
// packed bf16x2 + bf16x2 -> bf16x2 (f32 math)
__device__ __forceinline__ unsigned addpk(unsigned a, unsigned b) {
    float lo = bf2f((bf16_t)(a & 0xffffu)) + bf2f((bf16_t)(b & 0xffffu));
    float hi = bf2f((bf16_t)(a >> 16))     + bf2f((bf16_t)(b >> 16));
    return (unsigned)f2bf(lo) | ((unsigned)f2bf(hi) << 16);
}

__device__ __forceinline__ v8f wmma_bf16(const ABFrag& a, const ABFrag& b, v8f c) {
    return __builtin_amdgcn_wmma_f32_16x16x32_bf16(false, a.v, false, b.v,
                                                   (short)0, c, false, false);
}

// ---- one K=96 layer read from a wave-private LDS activation tile ----------
template<int NT>
__device__ __forceinline__ void layer_lds(const bf16_t* act,
                                          const unsigned* __restrict__ pw,
                                          v8f acc[NT], int lane)
{
    const int m  = lane & 15;
    const int kb = (lane >> 4) * 8;
    const unsigned* pwl = pw + lane * 8;
    #pragma unroll
    for (int t = 0; t < NT; ++t)
        #pragma unroll
        for (int q = 0; q < 8; ++q) acc[t][q] = 0.f;
    #pragma unroll
    for (int ks = 0; ks < 3; ++ks) {
        ABFrag a;
        const bf16_t* rp = act + m * ROWP + ks * 32 + kb;
        a.q[0] = *reinterpret_cast<const uint4*>(rp);
        a.q[1] = *reinterpret_cast<const uint4*>(rp + 16);
        ABFrag bf[NT];
        #pragma unroll
        for (int t = 0; t < NT; ++t) {
            const unsigned* wp = pwl + (size_t)(ks * NT + t) * 256;
            bf[t].q[0] = *reinterpret_cast<const uint4*>(wp);
            bf[t].q[1] = *reinterpret_cast<const uint4*>(wp + 4);
        }
        #pragma unroll
        for (int t = 0; t < NT; ++t) acc[t] = wmma_bf16(a, bf[t], acc[t]);
    }
}

// ---- apply bias (+optional ReLU) in-place to accumulators -----------------
template<int NT, bool RELU>
__device__ __forceinline__ void bias_relu(v8f acc[NT],
                                          const float* __restrict__ bias, int lane)
{
    const int n = lane & 15;
    #pragma unroll
    for (int t = 0; t < NT; ++t) {
        float bb = bias[t * 16 + n];
        #pragma unroll
        for (int r = 0; r < 8; ++r) {
            float x = acc[t][r] + bb;
            if (RELU) x = fmaxf(x, 0.f);
            acc[t][r] = x;
        }
    }
}

// ---- store D-fragments (bf16) into wave-private LDS tile ------------------
template<int NT>
__device__ __forceinline__ void store_tile(bf16_t* act, const v8f acc[NT], int lane)
{
    const int n     = lane & 15;
    const int rbase = (lane >> 4) * 8;
    #pragma unroll
    for (int t = 0; t < NT; ++t)
        #pragma unroll
        for (int r = 0; r < 8; ++r)
            act[(rbase + r) * ROWP + t * 16 + n] = f2bf(acc[t][r]);
}

// ---------------------------------------------------------------------------
// Fused 3-layer MLP:  out = L2( relu(L1( relu(L0(gathered_in)) )) )
//   NSEG==1 : layer0 K = K0STEPS*32 from one segment
//   NSEG>1  : NSEG segments of width 96 (K0STEPS == 3*NSEG)
//   layers 1,2: K = 96 from LDS. Hidden width fixed at 96 (6 N-tiles).
//   OUTF32 => final layer Nout=3 (one padded N-tile), f32 row-major out
//   else   => 96-wide bf16 result; optional fused epilogues:
//       out_bf16 != null : write result rows
//       agg/recv != null : atomicAdd f32 result into agg[recv[row]*96+col]
//       res      != null : res[row] = bf16(res[row] + result[row]) in-place
// ---------------------------------------------------------------------------
template<int NSEG, int K0STEPS, bool OUTF32>
__global__ __launch_bounds__(256)
void fused_mlp3(int R, Seg s0, Seg s1, Seg s2, Seg s3,
                const unsigned* __restrict__ pw0, const float* __restrict__ b0,
                const unsigned* __restrict__ pw1, const float* __restrict__ b1,
                const unsigned* __restrict__ pw2, const float* __restrict__ b2,
                bf16_t* __restrict__ out_bf16, float* __restrict__ out_f32,
                const int* __restrict__ recv, float* __restrict__ agg,
                bf16_t* __restrict__ res)
{
    constexpr int KPERSEG = (NSEG == 1) ? K0STEPS : 3;   // K-steps per segment
    static_assert(NSEG * KPERSEG == K0STEPS, "bad shape");

    __shared__ __align__(16) bf16_t smem[8 * 2 * 16 * ROWP];
    const int lane = threadIdx.x & 31;
    const int wave = threadIdx.x >> 5;
    const int rowbase = blockIdx.x * 128 + wave * 16;
    if (rowbase + 16 > R) return;                 // R % 16 == 0 always

    bf16_t* actA = smem + (size_t)(wave * 2 + 0) * 16 * ROWP;
    bf16_t* actB = smem + (size_t)(wave * 2 + 1) * 16 * ROWP;

    const int m    = lane & 15;
    const int kb   = (lane >> 4) * 8;
    const int grow = rowbase + m;

    // Per-segment A row pointers (constant indexing after unroll -> registers)
    const bf16_t* aptr[4];
    {
        const Seg ss[4] = { s0, s1, s2, s3 };
        #pragma unroll
        for (int s = 0; s < NSEG; ++s) {
            const int srow = ss[s].idx ? ss[s].idx[grow] : grow;
            aptr[s] = ss[s].base + (size_t)srow * (size_t)ss[s].stride + kb;
        }
    }

    // ---------------- layer 0 (global, multi-segment gather) ----------------
    v8f acc[6];
    #pragma unroll
    for (int t = 0; t < 6; ++t)
        #pragma unroll
        for (int q = 0; q < 8; ++q) acc[t][q] = 0.f;

    const unsigned* pw0l = pw0 + lane * 8;
    #pragma unroll
    for (int s = 0; s < NSEG; ++s) {
        #pragma unroll
        for (int k3 = 0; k3 < KPERSEG; ++k3) {
            const int ks   = s * KPERSEG + k3;
            const int kloc = k3 * 32;
            ABFrag a;
            a.q[0] = *reinterpret_cast<const uint4*>(aptr[s] + kloc);
            a.q[1] = *reinterpret_cast<const uint4*>(aptr[s] + kloc + 16);
            ABFrag bf[6];
            #pragma unroll
            for (int t = 0; t < 6; ++t) {
                const unsigned* wp = pw0l + (size_t)(ks * 6 + t) * 256;
                bf[t].q[0] = *reinterpret_cast<const uint4*>(wp);
                bf[t].q[1] = *reinterpret_cast<const uint4*>(wp + 4);
            }
            #pragma unroll
            for (int t = 0; t < 6; ++t) acc[t] = wmma_bf16(a, bf[t], acc[t]);
        }
    }
    bias_relu<6, true>(acc, b0, lane);
    store_tile<6>(actA, acc, lane);

    // ---------------- layer 1 -----------------------------------------------
    layer_lds<6>(actA, pw1, acc, lane);
    bias_relu<6, true>(acc, b1, lane);
    store_tile<6>(actB, acc, lane);

    // ---------------- layer 2 -----------------------------------------------
    if (OUTF32) {
        v8f acc1[1];
        layer_lds<1>(actB, pw2, acc1, lane);
        const int n     = lane & 15;
        const int rbase = (lane >> 4) * 8;
        if (n < 3) {
            float bb = b2[n];
            #pragma unroll
            for (int r = 0; r < 8; ++r)
                out_f32[(size_t)(rowbase + rbase + r) * 3 + n] = acc1[0][r] + bb;
        }
    } else {
        layer_lds<6>(actB, pw2, acc, lane);
        bias_relu<6, false>(acc, b2, lane);

        // fused segment_sum: agg[recv[row]*96+col] += result (f32, exact)
        if (agg) {
            const int n     = lane & 15;
            const int rbase = (lane >> 4) * 8;
            int rv[8];
            #pragma unroll
            for (int r = 0; r < 8; ++r) rv[r] = recv[rowbase + rbase + r];
            #pragma unroll
            for (int t = 0; t < 6; ++t)
                #pragma unroll
                for (int r = 0; r < 8; ++r)
                    atomicAdd(&agg[(size_t)rv[r] * 96 + t * 16 + n], acc[t][r]);
        }

        store_tile<6>(actA, acc, lane);
        // coalesced epilogue: 16 rows x 192B = 192 x 16B chunks, 6 per lane
        #pragma unroll
        for (int j = 0; j < 6; ++j) {
            int chunk = j * 32 + lane;
            int row   = chunk / 12;
            int c16   = chunk - row * 12;
            size_t go = (size_t)(rowbase + row) * 96 + c16 * 8;
            uint4 d = *reinterpret_cast<const uint4*>(actA + row * ROWP + c16 * 8);
            if (out_bf16) *reinterpret_cast<uint4*>(out_bf16 + go) = d;
            if (res) {      // residual in-place: res += result
                uint4 o = *reinterpret_cast<const uint4*>(res + go);
                uint4 u;
                u.x = addpk(o.x, d.x); u.y = addpk(o.y, d.y);
                u.z = addpk(o.z, d.z); u.w = addpk(o.w, d.w);
                *reinterpret_cast<uint4*>(res + go) = u;
            }
        }
    }
}

// ---------------------------------------------------------------------------
// Weight repack: f32 W[Ksrc x Nsrc] -> bf16 B-fragment layout, zero padded to
// Kpad x (NT*16).  out[((ks*NT+t)*32+lane)*8+j] packs K = ks*32+(lane>>4)*8+(j>>2)*16+2*(j&3)
// (lo half = even K), column = t*16 + (lane&15).
// ---------------------------------------------------------------------------
__global__ void pack_weights(const float* __restrict__ W, int Ksrc, int Nsrc,
                             int Kpad, int NT, unsigned* __restrict__ out)
{
    int idx = blockIdx.x * blockDim.x + threadIdx.x;
    int total = NT * (Kpad >> 5) * 256;
    if (idx >= total) return;
    int j    = idx & 7;
    int lane = (idx >> 3) & 31;
    int rest = idx >> 8;
    int t    = rest % NT;
    int ks   = rest / NT;
    int col  = t * 16 + (lane & 15);
    int k    = ks * 32 + (lane >> 4) * 8 + (j >> 2) * 16 + 2 * (j & 3);
    float f0 = (k     < Ksrc && col < Nsrc) ? W[(size_t)k       * Nsrc + col] : 0.f;
    float f1 = (k + 1 < Ksrc && col < Nsrc) ? W[(size_t)(k + 1) * Nsrc + col] : 0.f;
    out[idx] = (unsigned)f2bf(f0) | ((unsigned)f2bf(f1) << 16);
}

// f32[R x W] -> bf16[R x 32] zero-padded (for tiny encoder inputs)
__global__ void pad_convert(const float* __restrict__ in, int R, int W,
                            bf16_t* __restrict__ out)
{
    int i = blockIdx.x * blockDim.x + threadIdx.x;
    if (i >= R * 32) return;
    int r = i >> 5, c = i & 31;
    out[i] = (c < W) ? f2bf(in[(size_t)r * W + c]) : (bf16_t)0;
}

__global__ void zero_f32(float* __restrict__ p, int n)
{
    int i = blockIdx.x * blockDim.x + threadIdx.x;
    if (i < n) p[i] = 0.f;
}

__global__ void cvt_f32_bf16(const float* __restrict__ in,
                             bf16_t* __restrict__ out, int n)
{
    int i = blockIdx.x * blockDim.x + threadIdx.x;
    if (i < n) out[i] = f2bf(in[i]);
}

// column sums of bf16[R x 96] into f32[96] (one atomic per column per block)
__global__ void sum_rows(const bf16_t* __restrict__ x, int R,
                         float* __restrict__ out96)
{
    const int c  = threadIdx.x;           // 96 threads
    int r0 = blockIdx.x * 512;
    int r1 = r0 + 512; if (r1 > R) r1 = R;
    float s = 0.f;
    for (int r = r0; r < r1; ++r) s += bf2f(x[(size_t)r * 96 + c]);
    atomicAdd(&out96[c], s);
}

// column sums of f32[R x 96] into f32[96]
__global__ void sum_rows_f32(const float* __restrict__ x, int R,
                             float* __restrict__ out96)
{
    const int c  = threadIdx.x;           // 96 threads
    int r0 = blockIdx.x * 512;
    int r1 = r0 + 512; if (r1 > R) r1 = R;
    float s = 0.f;
    for (int r = r0; r < r1; ++r) s += x[(size_t)r * 96 + c];
    atomicAdd(&out96[c], s);
}

// tiny 3-layer MLP on one row (global-feature path); 96 threads, f32 math
__global__ void glob_mlp(const float* in0, int w0, const float* in1, int w1,
                         const float* in2, int w2,
                         const float* __restrict__ W0, const float* __restrict__ B0,
                         const float* __restrict__ W1, const float* __restrict__ B1,
                         const float* __restrict__ W2, const float* __restrict__ B2,
                         int nout, const float* resid,
                         float* out_f32, bf16_t* out_bf16)
{
    __shared__ float xin[288];
    __shared__ float h0[96];
    __shared__ float h1[96];
    const int t = threadIdx.x;
    const int K = w0 + w1 + w2;
    for (int k = t; k < K; k += 96) {
        float v;
        if      (k < w0)      v = in0[k];
        else if (k < w0 + w1) v = in1[k - w0];
        else                  v = in2[k - w0 - w1];
        xin[k] = v;
    }
    __syncthreads();
    float a = B0[t];
    for (int k = 0; k < K; ++k) a += xin[k] * W0[(size_t)k * 96 + t];
    h0[t] = fmaxf(a, 0.f);
    __syncthreads();
    float b = B1[t];
    for (int k = 0; k < 96; ++k) b += h0[k] * W1[(size_t)k * 96 + t];
    h1[t] = fmaxf(b, 0.f);
    __syncthreads();
    if (t < nout) {
        float c = B2[t];
        for (int k = 0; k < 96; ++k) c += h1[k] * W2[(size_t)k * nout + t];
        if (resid)    c += resid[t];
        if (out_f32)  out_f32[t]  = c;
        if (out_bf16) out_bf16[t] = f2bf(c);
    }
}

// ---------------------------------------------------------------------------
extern "C" void kernel_launch(void* const* d_in, const int* in_sizes, int n_in,
                              void* d_out, int out_size, void* d_ws, size_t ws_size,
                              hipStream_t stream)
{
    (void)in_sizes; (void)n_in; (void)out_size; (void)ws_size;

    const int*   edges     = (const int*)d_in[0];
    const int*   senders   = edges;
    const int*   receivers = edges + NE;
    const float* node_in   = (const float*)d_in[1];
    const float* edge_in   = (const float*)d_in[2];
    const float* glob_in   = (const float*)d_in[3];
    auto P = [&](int i) { return (const float*)d_in[i]; };

    // params pytree (sorted dict keys): dec_edge, dec_glob, dec_node,
    // enc_edge, enc_glob, enc_node, proc[ s ]{edge, glob, node}
    const int DEC_E = 4, DEC_G = 10, DEC_N = 16;
    const int ENC_E = 22, ENC_G = 28, ENC_N = 34, PROC0 = 40;

    // ---------------- workspace carve-up ------------------------------------
    char* wsb = (char*)d_ws;
    size_t off = 0;
    auto alloc = [&](size_t bytes) -> char* {
        char* p = wsb + off;
        off = (off + bytes + 255) & ~(size_t)255;
        return p;
    };
    unsigned* pwRegion = (unsigned*)alloc(2u << 20);
    bf16_t* node_pad = (bf16_t*)alloc((size_t)NN * 32 * 2);
    bf16_t* edge_pad = (bf16_t*)alloc((size_t)NE * 32 * 2);
    bf16_t* v   = (bf16_t*)alloc((size_t)NN * 96 * 2);
    bf16_t* e   = (bf16_t*)alloc((size_t)NE * 96 * 2);
    bf16_t* vn  = (bf16_t*)alloc((size_t)NN * 96 * 2);
    float*  aggf = (float*)alloc((size_t)NN * 96 * 4);
    bf16_t* aggb = (bf16_t*)alloc((size_t)NN * 96 * 2);
    float*  gA   = (float*)alloc(96 * 4);
    float*  gB   = (float*)alloc(96 * 4);
    bf16_t* gbf  = (bf16_t*)alloc(96 * 2);
    float*  sums = (float*)alloc(192 * 4);     // [0..95]=sum_v, [96..191]=sum_e

    // ---------------- weight pack table -------------------------------------
    struct PL { const float* W; const float* b; int Ksrc, Nsrc, Kpad, NT; unsigned* pw; };
    PL pls[36]; int np = 0; size_t pwOff = 0;
    auto addLayer = [&](const float* W, const float* b, int Ksrc, int Nsrc,
                        int Kpad, int NT) {
        PL& p = pls[np++];
        p.W = W; p.b = b; p.Ksrc = Ksrc; p.Nsrc = Nsrc; p.Kpad = Kpad; p.NT = NT;
        p.pw = (unsigned*)((char*)pwRegion + pwOff);
        size_t bytes = (size_t)NT * Kpad * 32;
        pwOff = (pwOff + bytes + 255) & ~(size_t)255;
    };
    auto addMLP3 = [&](int di, int K0src, int K0pad, int NsrcLast, int NTlast) {
        addLayer(P(di + 0), P(di + 1), K0src, 96, K0pad, 6);
        addLayer(P(di + 2), P(di + 3), 96, 96, 96, 6);
        addLayer(P(di + 4), P(di + 5), 96, NsrcLast, 96, NTlast);
    };
    addMLP3(ENC_N, 8, 32, 96, 6);     // 0..2
    addMLP3(ENC_E, 4, 32, 96, 6);     // 3..5
    addMLP3(DEC_N, 96, 96, 3, 1);     // 6..8
    addMLP3(DEC_E, 96, 96, 3, 1);     // 9..11
    for (int s = 0; s < 4; ++s) {
        addMLP3(PROC0 + 18 * s + 0,  384, 384, 96, 6);   // edge  -> 12+6s
        addMLP3(PROC0 + 18 * s + 12, 288, 288, 96, 6);   // node  -> 15+6s
    }

    for (int i = 0; i < np; ++i) {
        int total = pls[i].NT * (pls[i].Kpad / 32) * 256;
        pack_weights<<<(total + 255) / 256, 256, 0, stream>>>(
            pls[i].W, pls[i].Ksrc, pls[i].Nsrc, pls[i].Kpad, pls[i].NT, pls[i].pw);
    }

    // ---------------- input padding -----------------------------------------
    pad_convert<<<(NN * 32 + 255) / 256, 256, 0, stream>>>(node_in, NN, 8, node_pad);
    pad_convert<<<(NE * 32 + 255) / 256, 256, 0, stream>>>(edge_in, NE, 4, edge_pad);

    Seg Z{nullptr, nullptr, 0};

    // ---------------- encoders ----------------------------------------------
    fused_mlp3<1, 1, false><<<dim3((NN + 127) / 128), 256, 0, stream>>>(
        NN, Seg{node_pad, nullptr, 32}, Z, Z, Z,
        pls[0].pw, pls[0].b, pls[1].pw, pls[1].b, pls[2].pw, pls[2].b,
        v, nullptr, nullptr, nullptr, nullptr);
    fused_mlp3<1, 1, false><<<dim3((NE + 127) / 128), 256, 0, stream>>>(
        NE, Seg{edge_pad, nullptr, 32}, Z, Z, Z,
        pls[3].pw, pls[3].b, pls[4].pw, pls[4].b, pls[5].pw, pls[5].b,
        e, nullptr, nullptr, nullptr, nullptr);
    glob_mlp<<<1, 96, 0, stream>>>(glob_in, 4, nullptr, 0, nullptr, 0,
                                   P(ENC_G + 0), P(ENC_G + 1), P(ENC_G + 2),
                                   P(ENC_G + 3), P(ENC_G + 4), P(ENC_G + 5),
                                   96, nullptr, gA, gbf);
    float* gcur = gA; float* gnext = gB;

    // ---------------- processor (4 unshared GIN steps + residuals) ----------
    for (int s = 0; s < 4; ++s) {
        const int pb = PROC0 + 18 * s;
        const int le = 12 + 6 * s;   // edge MLP pack entries
        const int ln = 15 + 6 * s;   // node MLP pack entries
        // agg_e must be zero before the fused edge kernel's scatter
        zero_f32<<<(NN * 96 + 255) / 256, 256, 0, stream>>>(aggf, NN * 96);
        // e_new = MLP_edge([e || v_s || v_r || g]); fused:
        //   agg_e += segment_sum(e_new, receivers)   (f32 atomics)
        //   e     += e_new                           (in-place residual)
        fused_mlp3<4, 12, false><<<dim3((NE + 127) / 128), 256, 0, stream>>>(
            NE, Seg{e, nullptr, 96}, Seg{v, senders, 96},
            Seg{v, receivers, 96}, Seg{gbf, nullptr, 0},
            pls[le].pw, pls[le].b, pls[le + 1].pw, pls[le + 1].b,
            pls[le + 2].pw, pls[le + 2].b,
            nullptr, nullptr, receivers, aggf, e);
        cvt_f32_bf16<<<(NN * 96 + 255) / 256, 256, 0, stream>>>(aggf, aggb, NN * 96);
        // v_new = MLP_node([v || agg_e || g]); fused: v += v_new (in-place)
        fused_mlp3<3, 9, false><<<dim3((NN + 127) / 128), 256, 0, stream>>>(
            NN, Seg{v, nullptr, 96}, Seg{aggb, nullptr, 96},
            Seg{gbf, nullptr, 0}, Z,
            pls[ln].pw, pls[ln].b, pls[ln + 1].pw, pls[ln + 1].b,
            pls[ln + 2].pw, pls[ln + 2].b,
            vn, nullptr, nullptr, nullptr, v);
        // g_new = MLP_glob([g || sum(v_new) || sum(e_new)]);  g += g_new
        // (sum over agg_e == sum over e_new, exactly)
        zero_f32<<<1, 256, 0, stream>>>(sums, 192);
        sum_rows<<<(NN + 511) / 512, 96, 0, stream>>>(vn, NN, sums);
        sum_rows_f32<<<(NN + 511) / 512, 96, 0, stream>>>(aggf, NN, sums + 96);
        glob_mlp<<<1, 96, 0, stream>>>(gcur, 96, sums, 96, sums + 96, 96,
                                       P(pb + 6), P(pb + 7), P(pb + 8),
                                       P(pb + 9), P(pb + 10), P(pb + 11),
                                       96, gcur, gnext, gbf);
        { float* t = gcur; gcur = gnext; gnext = t; }
    }

    // ---------------- decoders ----------------------------------------------
    float* out = (float*)d_out;
    fused_mlp3<1, 3, true><<<dim3((NN + 127) / 128), 256, 0, stream>>>(
        NN, Seg{v, nullptr, 96}, Z, Z, Z,
        pls[6].pw, pls[6].b, pls[7].pw, pls[7].b, pls[8].pw, pls[8].b,
        nullptr, out, nullptr, nullptr, nullptr);
    fused_mlp3<1, 3, true><<<dim3((NE + 127) / 128), 256, 0, stream>>>(
        NE, Seg{e, nullptr, 96}, Z, Z, Z,
        pls[9].pw, pls[9].b, pls[10].pw, pls[10].b, pls[11].pw, pls[11].b,
        nullptr, out + (size_t)NN * 3, nullptr, nullptr, nullptr);
    glob_mlp<<<1, 96, 0, stream>>>(gcur, 96, nullptr, 0, nullptr, 0,
                                   P(DEC_G + 0), P(DEC_G + 1), P(DEC_G + 2),
                                   P(DEC_G + 3), P(DEC_G + 4), P(DEC_G + 5),
                                   3, nullptr, out + (size_t)NN * 3 + (size_t)NE * 3,
                                   nullptr);
}
